// Span_labler_6227702580112
// MI455X (gfx1250) — matrix operand order repaired and checked by reading, not compiled, not run on hardware
//
#include <hip/hip_runtime.h>
#include <math.h>

// ---------------- constants -------------------------------------------------
#define S_LEN   512
#define HID     512
#define NCAND   4096
#define WD      300
#define PD      50
#define DPD     50
#define POSD    5
#define IN0     405
#define IN0P    416            // padded to multiple of 32
#define L1IN    1024
#define G4H     2048           // 4*H
#define DSPAN   5170
#define DSPANP  5184           // padded to multiple of 32
#define H5      2560           // 5*H
#define LBL     70
#define LBLP    80             // padded to multiple of 16

#define BM      128
#define BN      128
#define BK      32

typedef __bf16 bf16;
typedef __attribute__((ext_vector_type(8)))  bf16  v8bf;
typedef __attribute__((ext_vector_type(16))) bf16  v16bf;
typedef __attribute__((ext_vector_type(8)))  float v8f;

__device__ __forceinline__ bf16 f2bf(float f) {
    unsigned u = __builtin_bit_cast(unsigned, f);
    unsigned r = u + 0x7fffu + ((u >> 16) & 1u);       // round-to-nearest-even
    unsigned short h = (unsigned short)(r >> 16);
    return __builtin_bit_cast(bf16, h);
}
__device__ __forceinline__ float sigm(float x) { return 1.0f / (1.0f + expf(-x)); }

__device__ __forceinline__ v16bf cat16(v8bf lo, v8bf hi) {
    return __builtin_shufflevector(lo, hi, 0,1,2,3,4,5,6,7,8,9,10,11,12,13,14,15);
}

// ---------------- 1) embedding gather -> X bf16 (512 x 416) ----------------
__global__ void embed_kernel(const int* __restrict__ word, const int* __restrict__ pos,
                             const int* __restrict__ dp, const int* __restrict__ argc,
                             const int* __restrict__ gold,
                             const float* __restrict__ wemb, const float* __restrict__ pemb,
                             const float* __restrict__ demb, const float* __restrict__ posemb,
                             bf16* __restrict__ X) {
    int s = blockIdx.x;
    int g = gold[0];
    int t0 = argc[2 * g], t1 = argc[2 * g + 1];
    int flag = (s >= t0 && s <= t1) ? 1 : 0;
    for (int c = threadIdx.x; c < IN0P; c += blockDim.x) {
        float v = 0.0f;
        if (c < WD)             v = wemb[(size_t)word[s] * WD + c];
        else if (c < WD+PD)     v = pemb[pos[s] * PD + (c - WD)];
        else if (c < WD+PD+DPD) v = demb[dp[s] * DPD + (c - WD - PD)];
        else if (c < IN0)       v = posemb[flag * POSD + (c - WD - PD - DPD)];
        X[(size_t)s * IN0P + c] = f2bf(v);
    }
}

// ---------------- f32 -> bf16 convert with zero padding ---------------------
__global__ void cvt_bf16_pad(const float* __restrict__ src, bf16* __restrict__ dst,
                             int rows, int cols, int cols_p) {
    int r = blockIdx.x;
    for (int c = threadIdx.x; c < cols_p; c += blockDim.x) {
        float v = (r < rows && c < cols) ? src[(size_t)r * cols + c] : 0.0f;
        dst[(size_t)r * cols_p + c] = f2bf(v);
    }
}

// ---------------- LDS-tiled WMMA GEMM: C(MxN) = A(MxK) * B(NxK)^T -----------
// 256 threads = 8 waves; 128x128 block tile; BK=32; double-buffered LDS staged
// with CDNA5 async-to-LDS loads (ASYNCcnt). M,N must be multiples of 128 and K
// a multiple of 32. Epilogue: +bias, optional ReLU, f32 and/or bf16 store.
__global__ __launch_bounds__(256) void gemm_wmma_tiled(
        const bf16* __restrict__ A, const bf16* __restrict__ B,
        const float* __restrict__ bias,
        float* __restrict__ Cf, bf16* __restrict__ Cbf,
        int K, int ldc, int relu) {
    __shared__ __align__(16) bf16 As[2][BM * BK];
    __shared__ __align__(16) bf16 Bs[2][BM * BK];

    const int tid  = threadIdx.x;          // 0..255
    const int m0   = blockIdx.x * BM;
    const int n0   = blockIdx.y * BN;
    const int w    = tid >> 5;             // wave 0..7 -> 2 (M) x 4 (N)
    const int lane = tid & 31;
    const int r    = lane & 15;
    const int kh   = lane >> 4;
    const int wm   = (w >> 2) * 64;        // wave M offset within tile
    const int wn   = (w & 3) * 32;         // wave N offset within tile

    // stage loader: 512 16-byte chunks per matrix tile, 2 per thread
    auto issue_stage = [&](int buf, int kk) {
        for (int c = tid; c < 512; c += 256) {
            int row = c >> 2, quad = c & 3;
            unsigned long long ga =
                (unsigned long long)(size_t)(A + (size_t)(m0 + row) * K + kk + quad * 8);
            unsigned la = (unsigned)(size_t)(&As[buf][row * BK + quad * 8]);
            asm volatile("global_load_async_to_lds_b128 %0, %1, off"
                         :: "v"(la), "v"(ga) : "memory");
            unsigned long long gb =
                (unsigned long long)(size_t)(B + (size_t)(n0 + row) * K + kk + quad * 8);
            unsigned lb = (unsigned)(size_t)(&Bs[buf][row * BK + quad * 8]);
            asm volatile("global_load_async_to_lds_b128 %0, %1, off"
                         :: "v"(lb), "v"(gb) : "memory");
        }
    };

    v8f zero = {};
    v8f acc[4][2];
    #pragma unroll
    for (int i = 0; i < 4; ++i)
        #pragma unroll
        for (int j = 0; j < 2; ++j) acc[i][j] = zero;

    issue_stage(0, 0);
    asm volatile("s_wait_asynccnt 0x0" ::: "memory");
    __syncthreads();

    int buf = 0;
    for (int kk = 0; kk < K; kk += BK) {
        if (kk + BK < K) issue_stage(buf ^ 1, kk + BK);

        const bf16* as = As[buf];
        const bf16* bs = Bs[buf];
        v16bf af[4], bfr[2];
        #pragma unroll
        for (int mt = 0; mt < 4; ++mt) {
            const bf16* p = as + (wm + mt * 16 + r) * BK + kh * 8;
            af[mt] = cat16(*(const v8bf*)p, *(const v8bf*)(p + 16));
        }
        #pragma unroll
        for (int nt = 0; nt < 2; ++nt) {
            const bf16* p = bs + (wn + nt * 16 + r) * BK + kh * 8;
            bfr[nt] = cat16(*(const v8bf*)p, *(const v8bf*)(p + 16));
        }
        #pragma unroll
        for (int mt = 0; mt < 4; ++mt)
            #pragma unroll
            for (int nt = 0; nt < 2; ++nt)
                acc[mt][nt] = __builtin_amdgcn_wmma_f32_16x16x32_bf16(
                    false, af[mt], false, bfr[nt], (short)0, acc[mt][nt], false, false);

        asm volatile("s_wait_asynccnt 0x0" ::: "memory");
        __syncthreads();
        buf ^= 1;
    }

    // epilogue
    #pragma unroll
    for (int nt = 0; nt < 2; ++nt) {
        const int col = n0 + wn + nt * 16 + r;
        const float bvv = bias ? bias[col] : 0.0f;
        #pragma unroll
        for (int mt = 0; mt < 4; ++mt) {
            const int mb = m0 + wm + mt * 16 + kh * 8;
            #pragma unroll
            for (int i = 0; i < 8; ++i) {
                float v = acc[mt][nt][i] + bvv;
                if (relu && v < 0.0f) v = 0.0f;
                if (Cf)  Cf [(size_t)(mb + i) * ldc + col] = v;
                if (Cbf) Cbf[(size_t)(mb + i) * ldc + col] = f2bf(v);
            }
        }
    }
}

// ---------------- simple per-wave WMMA GEMM (for small/padded-N GEMMs) ------
__global__ void gemm_wmma(const bf16* __restrict__ A, const bf16* __restrict__ B,
                          const float* __restrict__ bias,
                          float* __restrict__ Cf, bf16* __restrict__ Cbf,
                          int K, int ldc, int nreal, int relu) {
    const int m0 = blockIdx.x * 16;
    const int n0 = blockIdx.y * 16;
    const int lane = threadIdx.x;
    const int r  = lane & 15;
    const int kh = lane >> 4;

    const bf16* pa = A + (size_t)(m0 + r) * K + kh * 8;
    const bf16* pb = B + (size_t)(n0 + r) * K + kh * 8;

    v8f acc = {};
    for (int kk = 0; kk < K; kk += 32) {
        v8bf a0 = *(const v8bf*)(pa);
        v8bf a1 = *(const v8bf*)(pa + 16);
        v8bf b0 = *(const v8bf*)(pb);
        v8bf b1 = *(const v8bf*)(pb + 16);
        __builtin_prefetch(pa + 32, 0, 1);
        __builtin_prefetch(pb + 32, 0, 1);
        acc = __builtin_amdgcn_wmma_f32_16x16x32_bf16(false, cat16(a0, a1), false,
                                                      cat16(b0, b1), (short)0, acc,
                                                      false, false);
        pa += 32;
        pb += 32;
    }

    const int col = n0 + r;
    if (col < nreal) {
        const float bvv = bias ? bias[col] : 0.0f;
        const int mb = m0 + kh * 8;
        #pragma unroll
        for (int i = 0; i < 8; ++i) {
            float v = acc[i] + bvv;
            if (relu && v < 0.0f) v = 0.0f;
            if (Cf)  Cf [(size_t)(mb + i) * ldc + col] = v;
            if (Cbf) Cbf[(size_t)(mb + i) * ldc + col] = f2bf(v);
        }
    }
}

// ---------------- LSTM recurrence: one block per direction ------------------
__global__ void lstm_rec_kernel(const float* __restrict__ preF, const float* __restrict__ preB,
                                const float* __restrict__ Uf, const float* __restrict__ Ub,
                                float* __restrict__ xout) {
    const int dir = blockIdx.x;                 // 0 = forward, 1 = backward
    const float* pre = dir ? preB : preF;
    const float* U   = dir ? Ub   : Uf;
    const int tid = threadIdx.x;                // 0..1023

    __shared__ float sh[HID];
    __shared__ float sg[G4H];
    float c = 0.0f;
    if (tid < HID) sh[tid] = 0.0f;
    __syncthreads();

    const float* u0 = U + (size_t)tid * HID;
    const float* u1 = U + (size_t)(tid + 1024) * HID;

    for (int step = 0; step < S_LEN; ++step) {
        const int t = dir ? (S_LEN - 1 - step) : step;
        const float* prow = pre + (size_t)t * G4H;
        float a0 = prow[tid];
        float a1 = prow[tid + 1024];
        #pragma unroll 4
        for (int k = 0; k < HID; ++k) {
            float hk = sh[k];
            a0 += hk * u0[k];
            a1 += hk * u1[k];
        }
        sg[tid]        = a0;
        sg[tid + 1024] = a1;
        __syncthreads();
        if (tid < HID) {
            float ig = sigm(sg[tid]);
            float fg = sigm(sg[HID + tid]);
            float gg = tanhf(sg[2 * HID + tid]);
            float og = sigm(sg[3 * HID + tid]);
            c = fg * c + ig * gg;
            float hv = og * tanhf(c);
            sh[tid] = hv;
            xout[(size_t)t * (2 * HID) + dir * HID + tid] = hv;
        }
        __syncthreads();
    }
}

// ---------------- span feature gather -> feats bf16 (4096 x 5184) -----------
__global__ void build_feats(const float* __restrict__ x, const int* __restrict__ argc,
                            const int* __restrict__ head, const int* __restrict__ headdp,
                            const float* __restrict__ demb, bf16* __restrict__ feats) {
    const int i = blockIdx.x;
    const int a0 = argc[2 * i], a1 = argc[2 * i + 1];
    const int hd = head[i], hdpi = headdp[i];
    const float* xs = x + (size_t)a0 * L1IN;
    const float* xe = x + (size_t)a1 * L1IN;
    const float* xh = x + (size_t)hd * L1IN;
    for (int c = threadIdx.x; c < DSPANP; c += blockDim.x) {
        float v;
        if      (c < 1024) v = xs[c];
        else if (c < 2048) v = xe[c - 1024];
        else if (c < 3072) { int k = c - 2048; v = xs[k] + xe[k]; }
        else if (c < 4096) { int k = c - 3072; v = xs[k] - xe[k]; }
        else if (c < 5120) v = xh[c - 4096];
        else if (c < DSPAN) v = demb[hdpi * DPD + (c - 5120)];
        else v = 0.0f;
        feats[(size_t)i * DSPANP + c] = f2bf(v);
    }
}

// ---------------- log_softmax over axis 0 (per label column), in place ------
__global__ void logsoftmax_col(float* __restrict__ sc) {
    const int l = blockIdx.x;       // 0..69
    const int tid = threadIdx.x;    // 256 threads
    __shared__ float red[256];

    float m = -3.402823e38f;
    for (int i = tid; i < NCAND; i += 256) {
        float v = sc[(size_t)i * LBL + l];
        m = v > m ? v : m;
    }
    red[tid] = m;
    __syncthreads();
    for (int s = 128; s > 0; s >>= 1) {
        if (tid < s) red[tid] = fmaxf(red[tid], red[tid + s]);
        __syncthreads();
    }
    m = red[0];
    __syncthreads();

    float sum = 0.0f;
    for (int i = tid; i < NCAND; i += 256) sum += expf(sc[(size_t)i * LBL + l] - m);
    red[tid] = sum;
    __syncthreads();
    for (int s = 128; s > 0; s >>= 1) {
        if (tid < s) red[tid] += red[tid + s];
        __syncthreads();
    }
    const float lse = m + logf(red[0]);
    for (int i = tid; i < NCAND; i += 256) sc[(size_t)i * LBL + l] -= lse;
}

// ---------------- host orchestration ----------------------------------------
extern "C" void kernel_launch(void* const* d_in, const int* in_sizes, int n_in,
                              void* d_out, int out_size, void* d_ws, size_t ws_size,
                              hipStream_t stream) {
    (void)in_sizes; (void)n_in; (void)out_size; (void)ws_size;

    const int*   word   = (const int*)  d_in[0];
    const int*   pos    = (const int*)  d_in[1];
    const int*   dp     = (const int*)  d_in[2];
    const int*   argc   = (const int*)  d_in[3];
    const int*   head   = (const int*)  d_in[4];
    const int*   headdp = (const int*)  d_in[5];
    const int*   gold   = (const int*)  d_in[6];
    const float* wemb   = (const float*)d_in[7];
    const float* pemb   = (const float*)d_in[8];
    const float* demb   = (const float*)d_in[9];
    const float* posemb = (const float*)d_in[10];
    // lstm_params flattened: layer0 {Wf,Uf,bf,Wb,Ub,bb} = 11..16, layer1 = 17..22
    const float* W1 = (const float*)d_in[23];
    const float* b1 = (const float*)d_in[24];
    const float* W2 = (const float*)d_in[25];
    float* out = (float*)d_out;

    // workspace carve-up (all chunk sizes are multiples of 256 bytes)
    char* ws = (char*)d_ws;
    size_t off = 0;
    auto take = [&](size_t bytes) { void* p = ws + off; off += bytes; return p; };
    bf16*  Xbf   = (bf16*) take((size_t)S_LEN * L1IN * 2);        // 1 MB
    bf16*  Wfbf  = (bf16*) take((size_t)G4H * L1IN * 2);          // 4 MB
    bf16*  Wbbf  = (bf16*) take((size_t)G4H * L1IN * 2);          // 4 MB
    float* preF  = (float*)take((size_t)S_LEN * G4H * 4);         // 4 MB
    float* preB  = (float*)take((size_t)S_LEN * G4H * 4);         // 4 MB
    float* xnext = (float*)take((size_t)S_LEN * L1IN * 4);        // 2 MB
    bf16*  W1bf  = (bf16*) take((size_t)H5 * DSPANP * 2);         // 25.3 MB
    bf16*  feats = (bf16*) take((size_t)NCAND * DSPANP * 2);      // 40.5 MB
    bf16*  h1bf  = (bf16*) take((size_t)NCAND * H5 * 2);          // 20 MB
    bf16*  W2bf  = (bf16*) take((size_t)LBLP * H5 * 2);           // 0.4 MB

    // 1) embeddings -> X bf16 (512 x 416, zero padded)
    embed_kernel<<<S_LEN, 128, 0, stream>>>(word, pos, dp, argc, gold,
                                            wemb, pemb, demb, posemb, Xbf);

    // 2) BiLSTM layers
    for (int layer = 0; layer < 2; ++layer) {
        const int base = 11 + 6 * layer;
        const float* Wf = (const float*)d_in[base + 0];
        const float* Uf = (const float*)d_in[base + 1];
        const float* bf = (const float*)d_in[base + 2];
        const float* Wb = (const float*)d_in[base + 3];
        const float* Ub = (const float*)d_in[base + 4];
        const float* bb = (const float*)d_in[base + 5];
        const int in_dim  = layer == 0 ? IN0  : L1IN;
        const int in_dimp = layer == 0 ? IN0P : L1IN;

        cvt_bf16_pad<<<G4H, 256, 0, stream>>>(Wf, Wfbf, G4H, in_dim, in_dimp);
        cvt_bf16_pad<<<G4H, 256, 0, stream>>>(Wb, Wbbf, G4H, in_dim, in_dimp);

        dim3 g(S_LEN / BM, G4H / BN);       // (4, 16)
        gemm_wmma_tiled<<<g, 256, 0, stream>>>(Xbf, Wfbf, bf, preF, nullptr, in_dimp, G4H, 0);
        gemm_wmma_tiled<<<g, 256, 0, stream>>>(Xbf, Wbbf, bb, preB, nullptr, in_dimp, G4H, 0);

        lstm_rec_kernel<<<2, 1024, 0, stream>>>(preF, preB, Uf, Ub, xnext);

        if (layer == 0)
            cvt_bf16_pad<<<S_LEN, 256, 0, stream>>>(xnext, Xbf, S_LEN, L1IN, L1IN);
    }

    // 3) span FFN
    cvt_bf16_pad<<<H5, 256, 0, stream>>>(W1, W1bf, H5, DSPAN, DSPANP);
    build_feats<<<NCAND, 256, 0, stream>>>(xnext, argc, head, headdp, demb, feats);

    dim3 g1(NCAND / BM, H5 / BN);           // (32, 20): 108 GFLOP GEMM
    gemm_wmma_tiled<<<g1, 256, 0, stream>>>(feats, W1bf, b1, nullptr, h1bf, DSPANP, H5, 1);

    cvt_bf16_pad<<<LBLP, 256, 0, stream>>>(W2, W2bf, LBL, H5, H5);
    dim3 g2(NCAND / 16, LBLP / 16);         // score = h1 @ W2^T -> d_out (ldc = 70)
    gemm_wmma<<<g2, 32, 0, stream>>>(h1bf, W2bf, nullptr, out, nullptr, H5, LBL, LBL, 0);

    // 4) log_softmax across candidates (axis 0), in place on d_out
    logsoftmax_col<<<LBL, 256, 0, stream>>>(out);
}
